// ClientModel_68564857913800
// MI455X (gfx1250) — compile-verified, matrix operand order checked by reading, not run on hardware
//
#include <hip/hip_runtime.h>
#include <hip/hip_bf16.h>

#define SEQ 80
#define NC 80
#define EMBED 8
#define HID 256
#define BATCH 2048
#define FORGET_BIAS 1.0f

#define MTILE 16                    // batch rows per workgroup
#define HSTR 264                    // padded LDS row stride (bf16 elems): 528B, spreads banks
#define HB (MTILE * HSTR * 2)       // bytes per h buffer = 8448
#define NT_G 64                     // n-tiles in 4*HID = 1024
#define KT0 8                       // k-tiles layer0 (K = HID = 256)
#define KT1 16                      // k-tiles layer1 (K = 2*HID = 512)
#define KTD 8                       // k-tiles dense  (K = HID)
#define NTD 5                       // n-tiles dense  (80 classes)
#define PS 84                       // pred LDS row stride (floats)

#define FRAG_B   32                 // bytes per lane per fragment (16 bf16)
#define GATE_STR (16 * 32 * FRAG_B) // byte delta between gate n-tile groups = 16384
#define KT_STR   (NT_G * 32 * FRAG_B) // byte delta between k-tiles = 65536

// ---- workspace layout (bytes) ----
#define OFF_EMBPROJ 0
#define SZ_EMBPROJ  (NC * 4 * HID * 4)              // 327680
#define OFF_W0P     (OFF_EMBPROJ + SZ_EMBPROJ)
#define SZ_W0P      (KT0 * NT_G * 512 * 2)          // 524288
#define OFF_W1P     (OFF_W0P + SZ_W0P)
#define SZ_W1P      (KT1 * NT_G * 512 * 2)          // 1048576
#define OFF_WDP     (OFF_W1P + SZ_W1P)
#define SZ_WDP      (KTD * NTD * 512 * 2)           // 40960
#define OFF_PART    (OFF_WDP + SZ_WDP)              // 128 floats of partial losses

typedef __attribute__((ext_vector_type(16))) __bf16 v16bf;
typedef __attribute__((ext_vector_type(8)))  __bf16 v8bf;
typedef __attribute__((ext_vector_type(8)))  float  v8f;

#define WMMA_BF16(A, B, C) \
  __builtin_amdgcn_wmma_f32_16x16x32_bf16(false, (A), false, (B), (short)0, (C), false, false)

// fast sigmoid: v_exp_f32 + v_rcp_f32, no IEEE div expansion
__device__ __forceinline__ float sigmoidf_(float x) {
  return __builtin_amdgcn_rcpf(1.0f + __expf(-x));
}

// branch-free tanh: one v_exp_f32 + one v_rcp_f32, no divergence, overflow-safe
__device__ __forceinline__ float tanhf_(float x) {
  float a = fabsf(x);
  float e = __expf(-2.0f * a);          // in (0, 1]
  float t = (1.0f - e) * __builtin_amdgcn_rcpf(1.0f + e);
  return copysignf(t, x);
}

// A fragment (16x32 bf16, M x K) from LDS h buffer per the CDNA5 16-bit A layout:
// lanes 0-15 hold K = {k0..k0+7, k0+16..k0+23}; lanes 16-31 hold the +8 halves.
__device__ __forceinline__ v16bf load_afrag(const __bf16* hbuf, int lane, int k0) {
  int row = lane & 15;
  int kk  = k0 + ((lane & 16) ? 8 : 0);
  const __bf16* p = hbuf + row * HSTR + kk;
  v8bf lo = *(const v8bf*)p;
  v8bf hi = *(const v8bf*)(p + 16);
  v16bf a;
#pragma unroll
  for (int i = 0; i < 8; ++i) { a[i] = lo[i]; a[8 + i] = hi[i]; }
  return a;
}

// ---- prep: embproj[c][n] = b0[n] + sum_e emb[c][e] * W0[e][n] ----
__global__ void embproj_kernel(const float* __restrict__ emb,
                               const float* __restrict__ W0,
                               const float* __restrict__ b0,
                               float* __restrict__ out) {
  int idx = blockIdx.x * blockDim.x + threadIdx.x;
  if (idx >= NC * 4 * HID) return;
  int c = idx >> 10;
  int n = idx & 1023;
  float v = b0[n];
#pragma unroll
  for (int e = 0; e < EMBED; ++e)
    v += emb[c * EMBED + e] * W0[e * (4 * HID) + n];
  out[idx] = v;
}

// ---- prep: pack fp32 weights into bf16 WMMA B-fragment order ----
// frag(kt,nt) = 32 lanes x 16 bf16, lane-major; B (K x N) mirrors the A K-layout.
__global__ void pack_kernel(const float* __restrict__ src, int rowOff, int ld,
                            int NT, int total, __bf16* __restrict__ dst) {
  int idx = blockIdx.x * blockDim.x + threadIdx.x;
  if (idx >= total) return;
  int e    = idx & 15;
  int lane = (idx >> 4) & 31;
  int frag = idx >> 9;
  int kt = frag / NT;
  int nt = frag % NT;
  int kb = (lane & 16) ? 8 : 0;
  int k  = kt * 32 + ((e < 8) ? (kb + e) : (16 + kb + (e - 8)));
  int n  = nt * 16 + (lane & 15);
  dst[idx] = (__bf16)src[(rowOff + k) * ld + n];
}

// ---- main: one WG = 16 batch rows through all 80 timesteps ----
__global__ void lstm_kernel(const int* __restrict__ feats,
                            const int* __restrict__ labels,
                            const float* __restrict__ b1,
                            const float* __restrict__ bd,
                            const float* __restrict__ embproj,
                            const char* __restrict__ W0p,
                            const char* __restrict__ W1p,
                            const char* __restrict__ Wdp,
                            float* __restrict__ partials) {
  extern __shared__ char smem[];
  __bf16* h0b0 = (__bf16*)(smem);
  __bf16* h0b1 = (__bf16*)(smem + HB);
  __bf16* h1b0 = (__bf16*)(smem + 2 * HB);
  __bf16* h1b1 = (__bf16*)(smem + 3 * HB);

  const int tid  = threadIdx.x;
  const int lane = tid & 31;
  const int wave = tid >> 5;             // 8 waves; wave w owns gate-cols g = 2w, 2w+1
  const int col  = lane & 15;
  const int rhi  = (lane & 16) ? 8 : 0;  // C/D row offset for high half-wave
  const int wgBase = blockIdx.x * MTILE;

  // zero the t=0 "previous" h buffers
  {
    unsigned* z0 = (unsigned*)h0b1;
    unsigned* z1 = (unsigned*)h1b1;
    for (int i = tid; i < HB / 4; i += blockDim.x) { z0[i] = 0u; z1[i] = 0u; }
  }
  __syncthreads();

  v8f c0s[2], c1s[2];
#pragma unroll
  for (int g = 0; g < 2; ++g)
#pragma unroll
    for (int r = 0; r < 8; ++r) { c0s[g][r] = 0.0f; c1s[g][r] = 0.0f; }

  for (int t = 0; t < SEQ; ++t) {
    const int cur = t & 1;
    __bf16* h0prv = cur ? h0b0 : h0b1;
    __bf16* h0cur = cur ? h0b1 : h0b0;
    __bf16* h1prv = cur ? h1b0 : h1b1;
    __bf16* h1cur = cur ? h1b1 : h1b0;

    int ft[8];
#pragma unroll
    for (int r = 0; r < 8; ++r)
      ft[r] = feats[(wgBase + rhi + r) * SEQ + t];

    // ================= layer 0: gates = embproj[feat] + h0_prev @ W0h =================
#pragma unroll
    for (int gi = 0; gi < 2; ++gi) {
      const int g = wave * 2 + gi;
      // one base pointer; gates at constant immediate deltas, k-tiles at KT_STR stride
      const char* wbase = W0p + ((size_t)g * 32 + lane) * FRAG_B;
      v8f ai, aj, af, ao;
#pragma unroll
      for (int r = 0; r < 8; ++r) {
        const float* ep = embproj + ft[r] * (4 * HID) + g * 16 + col;
        ai[r] = ep[0];
        aj[r] = ep[HID];
        af[r] = ep[2 * HID];
        ao[r] = ep[3 * HID];
      }
      v16bf bi = *(const v16bf*)(wbase);
      v16bf bj = *(const v16bf*)(wbase + GATE_STR);
#pragma clang loop unroll(disable)
      for (int kk = 0; kk < KT0; ++kk) {
        const char* wk = wbase + (size_t)kk * KT_STR;
        v16bf bf = *(const v16bf*)(wk + 2 * GATE_STR);
        v16bf bo = *(const v16bf*)(wk + 3 * GATE_STR);
        v16bf a  = load_afrag(h0prv, lane, kk * 32);
        ai = WMMA_BF16(a, bi, ai);
        aj = WMMA_BF16(a, bj, aj);
        const char* wn = wbase + (size_t)((kk + 1 < KT0) ? kk + 1 : kk) * KT_STR;
        bi = *(const v16bf*)(wn);
        bj = *(const v16bf*)(wn + GATE_STR);
        af = WMMA_BF16(a, bf, af);
        ao = WMMA_BF16(a, bo, ao);
      }
#pragma unroll
      for (int r = 0; r < 8; ++r) {
        float c  = c0s[gi][r];
        float nc = c * sigmoidf_(af[r] + FORGET_BIAS) + sigmoidf_(ai[r]) * tanhf_(aj[r]);
        float nh = tanhf_(nc) * sigmoidf_(ao[r]);
        c0s[gi][r] = nc;
        h0cur[(rhi + r) * HSTR + g * 16 + col] = (__bf16)nh;
      }
    }
    __syncthreads();

    // ================= layer 1: gates = [h0_new, h1_prev] @ W1 + b1 =================
#pragma unroll
    for (int gi = 0; gi < 2; ++gi) {
      const int g = wave * 2 + gi;
      const char* wbase = W1p + ((size_t)g * 32 + lane) * FRAG_B;
      v8f ai = {}, aj = {}, af = {}, ao = {};
      auto loadA1 = [&](int kn) -> v16bf {
        return (kn < 8) ? load_afrag(h0cur, lane, kn * 32)
                        : load_afrag(h1prv, lane, (kn - 8) * 32);
      };
      v16bf bi = *(const v16bf*)(wbase);
      v16bf bj = *(const v16bf*)(wbase + GATE_STR);
#pragma clang loop unroll(disable)
      for (int kk = 0; kk < KT1; ++kk) {
        const char* wk = wbase + (size_t)kk * KT_STR;
        v16bf bf = *(const v16bf*)(wk + 2 * GATE_STR);
        v16bf bo = *(const v16bf*)(wk + 3 * GATE_STR);
        v16bf a  = loadA1(kk);
        ai = WMMA_BF16(a, bi, ai);
        aj = WMMA_BF16(a, bj, aj);
        const char* wn = wbase + (size_t)((kk + 1 < KT1) ? kk + 1 : kk) * KT_STR;
        bi = *(const v16bf*)(wn);
        bj = *(const v16bf*)(wn + GATE_STR);
        af = WMMA_BF16(a, bf, af);
        ao = WMMA_BF16(a, bo, ao);
      }
      const float bvi = b1[            g * 16 + col];
      const float bvj = b1[HID     +   g * 16 + col];
      const float bvf = b1[2 * HID +   g * 16 + col];
      const float bvo = b1[3 * HID +   g * 16 + col];
#pragma unroll
      for (int r = 0; r < 8; ++r) {
        float c  = c1s[gi][r];
        float nc = c * sigmoidf_(af[r] + bvf + FORGET_BIAS) +
                   sigmoidf_(ai[r] + bvi) * tanhf_(aj[r] + bvj);
        float nh = tanhf_(nc) * sigmoidf_(ao[r] + bvo);
        c1s[gi][r] = nc;
        h1cur[(rhi + r) * HSTR + g * 16 + col] = (__bf16)nh;
      }
    }
    __syncthreads();
  }

  // ---- dense head: pred = h1_final @ Wd + bd (h1 final buffer = h1b1, 79&1==1) ----
  float* predL = (float*)smem;              // [16][PS], reuses h0b0 region
  float* lossL = (float*)(smem + HB);       // [16], reuses h0b1 region
  const __bf16* h1fin = h1b1;
  if (wave < NTD) {
    const int nt = wave;
    const char* wdb = Wdp + ((size_t)nt * 32 + lane) * FRAG_B;
    v8f acc = {};
#pragma clang loop unroll(disable)
    for (int kk = 0; kk < KTD; ++kk) {
      v16bf a = load_afrag(h1fin, lane, kk * 32);
      v16bf b = *(const v16bf*)(wdb + (size_t)kk * (NTD * 32 * FRAG_B));
      acc = WMMA_BF16(a, b, acc);
    }
    const float bdv = bd[nt * 16 + col];
#pragma unroll
    for (int r = 0; r < 8; ++r)
      predL[(rhi + r) * PS + nt * 16 + col] = acc[r] + bdv;
  }
  __syncthreads();

  // ---- per-row log-softmax CE (deterministic, serial per row) ----
  if (tid < MTILE) {
    const int gm = wgBase + tid;
    const float* pr = predL + tid * PS;
    float mx = pr[0];
    for (int c = 1; c < NC; ++c) mx = fmaxf(mx, pr[c]);
    float se = 0.0f, pl = 0.0f;
    for (int c = 0; c < NC; ++c) {
      se += __expf(pr[c] - mx);
      if (labels[gm * NC + c] != 0) pl = pr[c];
    }
    lossL[tid] = (mx + __logf(se)) - pl;
  }
  __syncthreads();
  if (tid == 0) {
    float s = 0.0f;
    for (int r = 0; r < MTILE; ++r) s += lossL[r];
    partials[blockIdx.x] = s;
  }
}

// ---- deterministic final reduction (also overwrites poisoned d_out) ----
__global__ void finalize_kernel(const float* __restrict__ partials,
                                float* __restrict__ out) {
  if (threadIdx.x == 0 && blockIdx.x == 0) {
    float s = 0.0f;
    for (int i = 0; i < BATCH / MTILE; ++i) s += partials[i];
    out[0] = s / (float)BATCH;
  }
}

extern "C" void kernel_launch(void* const* d_in, const int* in_sizes, int n_in,
                              void* d_out, int out_size, void* d_ws, size_t ws_size,
                              hipStream_t stream) {
  (void)in_sizes; (void)n_in; (void)out_size; (void)ws_size;
  const int*   feats  = (const int*)d_in[0];
  const int*   labels = (const int*)d_in[1];
  const float* emb    = (const float*)d_in[2];
  const float* W0     = (const float*)d_in[3];
  const float* b0     = (const float*)d_in[4];
  const float* W1     = (const float*)d_in[5];
  const float* b1     = (const float*)d_in[6];
  const float* Wd     = (const float*)d_in[7];
  const float* bd     = (const float*)d_in[8];

  char* ws = (char*)d_ws;
  float*  embproj  = (float*)(ws + OFF_EMBPROJ);
  __bf16* W0p      = (__bf16*)(ws + OFF_W0P);
  __bf16* W1p      = (__bf16*)(ws + OFF_W1P);
  __bf16* Wdp      = (__bf16*)(ws + OFF_WDP);
  float*  partials = (float*)(ws + OFF_PART);

  {
    int total = NC * 4 * HID;
    embproj_kernel<<<(total + 255) / 256, 256, 0, stream>>>(emb, W0, b0, embproj);
  }
  {
    int total = KT0 * NT_G * 512;
    pack_kernel<<<(total + 255) / 256, 256, 0, stream>>>(W0, EMBED, 4 * HID, NT_G, total, W0p);
  }
  {
    int total = KT1 * NT_G * 512;
    pack_kernel<<<(total + 255) / 256, 256, 0, stream>>>(W1, 0, 4 * HID, NT_G, total, W1p);
  }
  {
    int total = KTD * NTD * 512;
    pack_kernel<<<(total + 255) / 256, 256, 0, stream>>>(Wd, 0, NC, NTD, total, Wdp);
  }
  lstm_kernel<<<BATCH / MTILE, 256, 4 * HB, stream>>>(
      feats, labels, b1, bd, embproj,
      (const char*)W0p, (const char*)W1p, (const char*)Wdp, partials);
  finalize_kernel<<<1, 32, 0, stream>>>(partials, (float*)d_out);
}
